// GPSA_26912265077208
// MI455X (gfx1250) — matrix operand order repaired and checked by reading, not compiled, not run on hardware
//
#include <hip/hip_runtime.h>

typedef __bf16 bf16;
typedef __attribute__((ext_vector_type(16))) __bf16 v16bf;
typedef __attribute__((ext_vector_type(8)))  __bf16 v8bf;
typedef __attribute__((ext_vector_type(8)))  float  v8f;

#define DIMC  256
#define NHEAD 8
#define HD    32
#define NTOK  1024
#define BATCH 8

__device__ __forceinline__ v8f vzero8() {
  v8f z;
#pragma unroll
  for (int i = 0; i < 8; ++i) z[i] = 0.0f;
  return z;
}

__device__ __forceinline__ v16bf cat8(v8bf lo, v8bf hi) {
  return __builtin_shufflevector(lo, hi, 0,1,2,3,4,5,6,7,8,9,10,11,12,13,14,15);
}

// A-matrix 16x32 bf16 tile (ISA 7.12.2): lane m = lane&15, h = lane>>4.
// element e=0..7  -> K = h*8 + e      (VGPR 0..3)
// element e=8..15 -> K = 16 + h*8 + (e-8)  (VGPR 4..7)
// rowm points at this lane's row (row-major, K contiguous), at k0.
__device__ __forceinline__ v16bf load_tileA(const bf16* rowm, int h) {
  v8bf lo = *(const v8bf*)(rowm + h * 8);
  v8bf hi = *(const v8bf*)(rowm + 16 + h * 8);
  return cat8(lo, hi);
}

// B-matrix 32x16 bf16 tile: lane n = lane&15, element e=0..7 -> K = h*8 + e, e=8..15 -> K = 16 + h*8 + (e-8).
// Source stored [n][k] (K contiguous per output column); rown at column row, k0.
__device__ __forceinline__ v16bf load_tileB(const bf16* rown, int h) {
  v8bf lo = *(const v8bf*)(rown + h * 8);
  v8bf hi = *(const v8bf*)(rown + 16 + h * 8);
  return cat8(lo, hi);
}

__device__ __forceinline__ v8f wmma_bf16(v16bf a, v16bf b, v8f c) {
  return __builtin_amdgcn_wmma_f32_16x16x32_bf16(false, a, false, b, (short)0, c,
                                                 false, false);
}

// ---------------- pack weights fp32 -> bf16 ----------------
__global__ void k_pack_weights(const float* __restrict__ wqk,
                               const float* __restrict__ wv,
                               const float* __restrict__ wproj,
                               bf16* __restrict__ wqk_b,
                               bf16* __restrict__ wv_b,
                               bf16* __restrict__ wproj_b) {
  int i = blockIdx.x * 256 + threadIdx.x;
  if (i < 512 * 256) wqk_b[i] = (bf16)wqk[i];
  if (i < 256 * 256) {
    wv_b[i]    = (bf16)wv[i];
    wproj_b[i] = (bf16)wproj[i];
  }
}

// ---------------- x (B,C,N) -> xf (B,N,C) bf16 ----------------
__global__ void k_pack_x(const float* __restrict__ x, bf16* __restrict__ xf) {
  int idx = blockIdx.x * 256 + threadIdx.x;          // B*N*C = 2M
  int c = idx & 255;
  int n = (idx >> 8) & 1023;
  int b = idx >> 18;
  xf[idx] = (bf16)x[((size_t)b * DIMC + c) * NTOK + n];
}

// ---------------- QKV GEMM: one wave = one 16x16 output tile ----------------
// grid.x = 512 (b,n tiles), grid.y = 48 (0..31 -> Wqk cols, 32..47 -> Wv cols)
__global__ void k_qkv(const bf16* __restrict__ xf, const bf16* __restrict__ wqk,
                      const bf16* __restrict__ wv,
                      bf16* __restrict__ Q, bf16* __restrict__ Kt,
                      bf16* __restrict__ VT) {
  const int lane = threadIdx.x, h = lane >> 4, lm = lane & 15;
  const int mt = blockIdx.x, ct = blockIdx.y;
  const int b = mt >> 6, n0 = (mt & 63) << 4;
  const bf16* arow = xf + ((size_t)b * NTOK + n0 + lm) * DIMC;
  const bool isV = ct >= 32;
  const bf16* W = isV ? wv : wqk;
  const int col0 = (isV ? ct - 32 : ct) << 4;
  const bf16* brow = W + (size_t)(col0 + lm) * DIMC;

  v8f acc = vzero8();
#pragma unroll
  for (int k0 = 0; k0 < DIMC; k0 += 32)
    acc = wmma_bf16(load_tileA(arow + k0, h), load_tileB(brow + k0, h), acc);

#pragma unroll
  for (int r = 0; r < 8; ++r) {
    int n = n0 + h * 8 + r;                          // C layout: M = r + 8*(lane>=16)
    float v = acc[r];
    if (!isV) {
      int c = ct * 16 + lm;                          // 0..511 over Wqk rows
      int cc = c & 255, hh = cc >> 5, d = cc & 31;
      size_t o = (((size_t)b * NHEAD + hh) * NTOK + n) * HD + d;
      if (c < 256) Q[o] = (bf16)v; else Kt[o] = (bf16)v;
    } else {
      int c = col0 + lm, hh = c >> 5, d = c & 31;
      VT[(((size_t)b * NHEAD + hh) * HD + d) * NTOK + n] = (bf16)v;   // V^T [d][n]
    }
  }
}

// ---------------- position softmax stats + exp table (batch-independent) ----
__global__ void k_pos(const float* __restrict__ Wpos, const float* __restrict__ bpos,
                      float* __restrict__ Mq, float* __restrict__ Lq,
                      bf16* __restrict__ posw) {
  int tid = blockIdx.x * 256 + threadIdx.x;          // NHEAD*NTOK = 8192 rows
  int hh = tid >> 10, n = tid & 1023;
  float w0 = Wpos[hh * 3 + 0], w1 = Wpos[hh * 3 + 1], w2 = Wpos[hh * 3 + 2];
  float bp = bpos[hh];
  int nx = n & 31, ny = n >> 5;
  float mx = -3.4e38f, sm = 0.0f;
  for (int m = 0; m < NTOK; ++m) {
    float dx = (float)((m & 31) - nx), dy = (float)((m >> 5) - ny);
    float t = w0 * dx + w1 * dy + w2 * (dx * dx + dy * dy) + bp;
    float m2 = fmaxf(mx, t);
    sm = sm * __expf(mx - m2) + __expf(t - m2);
    mx = m2;
  }
  Mq[tid] = mx;
  Lq[tid] = sm;
  bf16* pr = posw + (size_t)tid * NTOK;
  for (int m = 0; m < NTOK; ++m) {
    float dx = (float)((m & 31) - nx), dy = (float)((m >> 5) - ny);
    float t = w0 * dx + w1 * dy + w2 * (dx * dx + dy * dy) + bp;
    pr[m] = (bf16)__expf(t - mx);
  }
}

// ---------------- patch softmax stats: S = Q K^T * scale via WMMA ------------
// grid.x = B*NH*64; one wave per 16-query tile, sweep 64 key tiles.
__global__ void k_patch_stats(const bf16* __restrict__ Q, const bf16* __restrict__ Kt,
                              float* __restrict__ Mp, float* __restrict__ Lp) {
  const int lane = threadIdx.x, h = lane >> 4, lm = lane & 15;
  const int bh = blockIdx.x >> 6, n0 = (blockIdx.x & 63) << 4;
  const float scale = 0.17677669529663687f;          // hd^-0.5
  v16bf a = load_tileA(Q + ((size_t)bh * NTOK + n0 + lm) * HD, h);
  float mx[8], sm[8];
#pragma unroll
  for (int r = 0; r < 8; ++r) { mx[r] = -3.4e38f; sm[r] = 0.0f; }
  for (int kt = 0; kt < 64; ++kt) {
    v16bf bm = load_tileB(Kt + ((size_t)bh * NTOK + kt * 16 + lm) * HD, h);
    v8f s = wmma_bf16(a, bm, vzero8());
#pragma unroll
    for (int r = 0; r < 8; ++r) {
      float t = s[r] * scale;
      float m2 = fmaxf(mx[r], t);
      sm[r] = sm[r] * __expf(mx[r] - m2) + __expf(t - m2);
      mx[r] = m2;
    }
  }
#pragma unroll
  for (int r = 0; r < 8; ++r) {
    for (int off = 1; off < 16; off <<= 1) {         // reduce within 16-lane half
      float om = __shfl_xor(mx[r], off, 32);
      float os = __shfl_xor(sm[r], off, 32);
      float m2 = fmaxf(mx[r], om);
      sm[r] = sm[r] * __expf(mx[r] - m2) + os * __expf(om - m2);
      mx[r] = m2;
    }
    if (lm == 0) {
      int n = n0 + h * 8 + r;
      Mp[(size_t)bh * NTOK + n] = mx[r];
      Lp[(size_t)bh * NTOK + n] = sm[r];
    }
  }
}

// ---------------- fused gated attention: recompute S, combine, attn @ V -----
// out_row = [(1-g)/Lp * sum(exp(s-Mp) m v) + g/Lq * sum(posw m v)] / den
// den     = (1-g)/Lp * sum(exp(s-Mp) m)    + g/Lq * sum(posw m)
__global__ void k_attn(const bf16* __restrict__ Q, const bf16* __restrict__ Kt,
                       const bf16* __restrict__ VT, const bf16* __restrict__ posw,
                       const float* __restrict__ Mp, const float* __restrict__ Lp,
                       const float* __restrict__ Mq, const float* __restrict__ Lq,
                       const float* __restrict__ mask, const float* __restrict__ gating,
                       bf16* __restrict__ O) {
  __shared__ bf16 lds[16 * 32];                      // exp-weight tile, A layout
  const int lane = threadIdx.x, h = lane >> 4, lm = lane & 15;
  const int bh = blockIdx.x >> 6, b = bh >> 3, hh = bh & 7;
  const int n0 = (blockIdx.x & 63) << 4;
  const float scale = 0.17677669529663687f;
  const float g = 1.0f / (1.0f + __expf(-gating[hh]));

  v16bf a = load_tileA(Q + ((size_t)bh * NTOK + n0 + lm) * HD, h);
  float mp[8], cp[8], mq[8], cq[8], den[8];
#pragma unroll
  for (int r = 0; r < 8; ++r) {
    int n = n0 + h * 8 + r;
    mp[r] = Mp[(size_t)bh * NTOK + n];
    cp[r] = (1.0f - g) / Lp[(size_t)bh * NTOK + n];
    mq[r] = Mq[(size_t)hh * NTOK + n];
    cq[r] = g / Lq[(size_t)hh * NTOK + n];
    den[r] = 0.0f;
  }
  v8f o0 = vzero8(), o1 = vzero8();

  for (int ks = 0; ks < 32; ++ks) {                  // 32 keys per step
#pragma unroll
    for (int sub = 0; sub < 2; ++sub) {
      int kt = ks * 2 + sub;
      v16bf bm = load_tileB(Kt + ((size_t)bh * NTOK + kt * 16 + lm) * HD, h);
      v8f s = wmma_bf16(a, bm, vzero8());
      int key = kt * 16 + lm;
      float mk = mask[b * NTOK + key];
#pragma unroll
      for (int r = 0; r < 8; ++r) {
        int n = n0 + h * 8 + r;
        float pw = (float)posw[((size_t)hh * NTOK + n) * NTOK + key]; // L2-resident
        float w = cp[r] * __expf(s[r] * scale - mp[r]) + cq[r] * pw;
        w *= mk;
        den[r] += w;
        lds[(h * 8 + r) * 32 + sub * 16 + lm] = (bf16)w;
      }
    }
    __syncthreads();
    v16bf pa = load_tileA(lds + lm * 32, h);
    v16bf b0 = load_tileB(VT + ((size_t)bh * HD + lm)      * NTOK + ks * 32, h);
    v16bf b1 = load_tileB(VT + ((size_t)bh * HD + 16 + lm) * NTOK + ks * 32, h);
    o0 = wmma_bf16(pa, b0, o0);
    o1 = wmma_bf16(pa, b1, o1);
    __syncthreads();
  }

#pragma unroll
  for (int r = 0; r < 8; ++r) {
    for (int off = 1; off < 16; off <<= 1) den[r] += __shfl_xor(den[r], off, 32);
    float inv = 1.0f / den[r];
    int n = n0 + h * 8 + r;
    bf16* orow = O + ((size_t)b * NTOK + n) * DIMC + hh * HD;
    orow[lm]      = (bf16)(o0[r] * inv);
    orow[16 + lm] = (bf16)(o1[r] * inv);
  }
}

// ---------------- output projection + bias + transpose to (B,C,H,W) ---------
__global__ void k_proj(const bf16* __restrict__ O, const bf16* __restrict__ wproj,
                       const float* __restrict__ bproj, float* __restrict__ out) {
  const int lane = threadIdx.x, h = lane >> 4, lm = lane & 15;
  const int mt = blockIdx.x, ct = blockIdx.y;
  const int b = mt >> 6, n0 = (mt & 63) << 4;
  const bf16* arow = O + ((size_t)b * NTOK + n0 + lm) * DIMC;
  const bf16* brow = wproj + (size_t)(ct * 16 + lm) * DIMC;
  v8f acc = vzero8();
#pragma unroll
  for (int k0 = 0; k0 < DIMC; k0 += 32)
    acc = wmma_bf16(load_tileA(arow + k0, h), load_tileB(brow + k0, h), acc);
  int c = ct * 16 + lm;
  float bias = bproj[c];
#pragma unroll
  for (int r = 0; r < 8; ++r) {
    int n = n0 + h * 8 + r;
    out[((size_t)b * DIMC + c) * NTOK + n] = acc[r] + bias;
  }
}

extern "C" void kernel_launch(void* const* d_in, const int* in_sizes, int n_in,
                              void* d_out, int out_size, void* d_ws, size_t ws_size,
                              hipStream_t stream) {
  const float* x      = (const float*)d_in[0];
  const float* maskp  = (const float*)d_in[1];
  const float* wqk    = (const float*)d_in[2];
  const float* wv     = (const float*)d_in[3];
  const float* wproj  = (const float*)d_in[4];
  const float* bproj  = (const float*)d_in[5];
  const float* wpos   = (const float*)d_in[6];
  const float* bpos   = (const float*)d_in[7];
  const float* gating = (const float*)d_in[8];
  float* out = (float*)d_out;

  char* ws = (char*)d_ws;
  auto alloc = [&](size_t bytes) -> void* {
    void* p = (void*)ws;
    ws += (bytes + 255) & ~(size_t)255;
    return p;
  };
  bf16* xf      = (bf16*)alloc((size_t)BATCH * NTOK * DIMC * 2);
  bf16* wqk_b   = (bf16*)alloc((size_t)512 * 256 * 2);
  bf16* wv_b    = (bf16*)alloc((size_t)256 * 256 * 2);
  bf16* wproj_b = (bf16*)alloc((size_t)256 * 256 * 2);
  bf16* Qb      = (bf16*)alloc((size_t)BATCH * NHEAD * NTOK * HD * 2);
  bf16* Kb      = (bf16*)alloc((size_t)BATCH * NHEAD * NTOK * HD * 2);
  bf16* VTb     = (bf16*)alloc((size_t)BATCH * NHEAD * NTOK * HD * 2);
  bf16* Ob      = (bf16*)alloc((size_t)BATCH * NTOK * DIMC * 2);
  float* Mp     = (float*)alloc((size_t)BATCH * NHEAD * NTOK * 4);
  float* Lp     = (float*)alloc((size_t)BATCH * NHEAD * NTOK * 4);
  float* Mq     = (float*)alloc((size_t)NHEAD * NTOK * 4);
  float* Lq     = (float*)alloc((size_t)NHEAD * NTOK * 4);
  bf16* posw    = (bf16*)alloc((size_t)NHEAD * NTOK * NTOK * 2);

  k_pack_weights<<<512, 256, 0, stream>>>(wqk, wv, wproj, wqk_b, wv_b, wproj_b);
  k_pack_x<<<(BATCH * NTOK * DIMC) / 256, 256, 0, stream>>>(x, xf);
  k_qkv<<<dim3(512, 48), 32, 0, stream>>>(xf, wqk_b, wv_b, Qb, Kb, VTb);
  k_pos<<<(NHEAD * NTOK) / 256, 256, 0, stream>>>(wpos, bpos, Mq, Lq, posw);
  k_patch_stats<<<BATCH * NHEAD * 64, 32, 0, stream>>>(Qb, Kb, Mp, Lp);
  k_attn<<<BATCH * NHEAD * 64, 32, 0, stream>>>(Qb, Kb, VTb, posw, Mp, Lp, Mq, Lq,
                                                maskp, gating, Ob);
  k_proj<<<dim3(512, 16), 32, 0, stream>>>(Ob, wproj_b, bproj, out);
}
